// MultiHeadAttention_23527830847569
// MI455X (gfx1250) — compile-verified, matrix operand order checked by reading, not run on hardware
//
#include <hip/hip_runtime.h>
#include <hip/hip_bf16.h>

// ---------------------------------------------------------------------------
// Types for CDNA5 WMMA (wave32): D(16x16 f32) = A(16x32 bf16) x B(32x16 bf16) + C
// ---------------------------------------------------------------------------
typedef __attribute__((ext_vector_type(16))) __bf16 v16bf;
typedef __attribute__((ext_vector_type(8)))  __bf16 v8bf;
typedef __attribute__((ext_vector_type(4)))  __bf16 v4bf;
typedef __attribute__((ext_vector_type(8)))  float  v8f;
typedef __attribute__((ext_vector_type(4)))  float  v4f;

#define B_   2
#define S_   2048
#define H_   16
#define DK_  64
#define DM_  1024
#define MROWS_ (B_ * S_)   // 4096

__device__ inline v8f wmma_bf16(v16bf a, v16bf b, v8f c) {
  return __builtin_amdgcn_wmma_f32_16x16x32_bf16(false, a, false, b, (short)0, c,
                                                 false, false);
}

__device__ inline v8f zero8() {
  v8f z;
#pragma unroll
  for (int i = 0; i < 8; ++i) z[i] = 0.0f;
  return z;
}

// Load one 16x32 bf16 fragment in the ISA-documented 16-bit A/B VGPR layout.
// src is row-major with leading dim `ld` (halves). For A-fragments rows are the
// 16 M-rows; for B-fragments the "rows" are the 16 N-columns of a pre-transposed
// (NxK) operand. Per lane: half j -> K = 16*(j/8) + 8*(lane/16) + (j%8), i.e.
// two contiguous 16-byte chunks at kb + 8*half and kb + 16 + 8*half.
__device__ inline v16bf load_frag(const __bf16* __restrict__ src, int ld,
                                  int row0, int kb) {
  int lane = threadIdx.x & 31;
  int m    = lane & 15;
  int half = lane >> 4;
  const __bf16* p = src + (size_t)(row0 + m) * ld + kb + half * 8;
  v8bf lo = *reinterpret_cast<const v8bf*>(p);
  v8bf hi = *reinterpret_cast<const v8bf*>(p + 16);
  return __builtin_shufflevector(lo, hi, 0, 1, 2, 3, 4, 5, 6, 7,
                                 8, 9, 10, 11, 12, 13, 14, 15);
}

// ---------------------------------------------------------------------------
// fp32 -> bf16 pack (4 elements / thread)
// ---------------------------------------------------------------------------
__global__ void cvt_f32_bf16(const float* __restrict__ src,
                             __bf16* __restrict__ dst, int n4) {
  int i = blockIdx.x * blockDim.x + threadIdx.x;
  if (i < n4) {
    v4f x = reinterpret_cast<const v4f*>(src)[i];
    reinterpret_cast<v4bf*>(dst)[i] = __builtin_convertvector(x, v4bf);
  }
}

// ---------------------------------------------------------------------------
// W [K=1024][N=1024] fp32 -> WT [N][K] bf16 (LDS-tiled transpose)
// ---------------------------------------------------------------------------
__global__ void transpose_w_bf16(const float* __restrict__ W,
                                 __bf16* __restrict__ WT) {
  __shared__ float tile[32][33];
  int x  = blockIdx.x * 32 + threadIdx.x;   // n
  int y0 = blockIdx.y * 32 + threadIdx.y;   // k
#pragma unroll
  for (int i = 0; i < 32; i += 8)
    tile[threadIdx.y + i][threadIdx.x] = W[(size_t)(y0 + i) * DM_ + x];
  __syncthreads();
  int nx  = blockIdx.y * 32 + threadIdx.x;  // k
  int ny0 = blockIdx.x * 32 + threadIdx.y;  // n
#pragma unroll
  for (int i = 0; i < 32; i += 8)
    WT[(size_t)(ny0 + i) * DM_ + nx] = (__bf16)tile[threadIdx.x][threadIdx.y + i];
}

// ---------------------------------------------------------------------------
// lens[0..1] = q_len[b], lens[2..3] = k_len[b]
// ---------------------------------------------------------------------------
__global__ void compute_lens(const int* __restrict__ qmask,
                             const int* __restrict__ kmask,
                             int* __restrict__ lens) {
  __shared__ int sm[256];
  int b = blockIdx.x;
  const int* src = (blockIdx.y == 0) ? qmask : kmask;
  int s = 0;
  for (int i = threadIdx.x; i < S_; i += 256) s += src[b * S_ + i];
  sm[threadIdx.x] = s;
  __syncthreads();
  for (int off = 128; off > 0; off >>= 1) {
    if ((int)threadIdx.x < off) sm[threadIdx.x] += sm[threadIdx.x + off];
    __syncthreads();
  }
  if (threadIdx.x == 0) lens[blockIdx.y * 2 + b] = sm[0];
}

// ---------------------------------------------------------------------------
// GEMM: out[4096 x 1024] = X(bf16) @ WT^T(bf16) + bias, WMMA bf16, f32 acc.
// Each wave: 32x64 tile (2 M-subtiles x 4 N-subtiles = 8 WMMAs per K-step,
// B-fragments shared across M). K-loop is software-pipelined with ping-pong
// register buffers (two 32-wide K-steps per iteration, branch-free steady
// state + peeled epilogue). __launch_bounds__(256,2) raises the VGPR budget
// so the ~200-VGPR live set stays in registers (round-2 build spilled).
// mode 0: Q head-split bf16, * 1/sqrt(64)   [B,H,S,64]
// mode 1: K head-split bf16                  [B,H,S,64]
// mode 2: V head-split transposed bf16       [B,H,64,S]
// mode 3: fp32 dense output                  [B*S,1024]
// ---------------------------------------------------------------------------
__global__ __launch_bounds__(256, 2)
void gemm_bf16(const __bf16* __restrict__ X,
               const __bf16* __restrict__ WT,
               const float* __restrict__ bias,
               void* __restrict__ dst, int mode) {
  int w  = threadIdx.x >> 5;
  int m0 = blockIdx.x * 256 + w * 32;
  int n0 = blockIdx.y * 64;

  v8f acc[2][4];
#pragma unroll
  for (int mi = 0; mi < 2; ++mi)
#pragma unroll
    for (int t = 0; t < 4; ++t) acc[mi][t] = zero8();

  v16bf a0[2], b0[4], a1[2], b1[4];

  // prologue: load K-step 0 into buffer 0
#pragma unroll
  for (int mi = 0; mi < 2; ++mi) a0[mi] = load_frag(X, DM_, m0 + mi * 16, 0);
#pragma unroll
  for (int t = 0; t < 4; ++t) b0[t] = load_frag(WT, DM_, n0 + t * 16, 0);

  // steady state: two 32-wide K-steps per iteration, branch-free prefetch
  int kb = 0;
  for (; kb < DM_ - 64; kb += 64) {
#pragma unroll
    for (int mi = 0; mi < 2; ++mi)
      a1[mi] = load_frag(X, DM_, m0 + mi * 16, kb + 32);
#pragma unroll
    for (int t = 0; t < 4; ++t)
      b1[t] = load_frag(WT, DM_, n0 + t * 16, kb + 32);

#pragma unroll
    for (int t = 0; t < 4; ++t) {
      acc[0][t] = wmma_bf16(a0[0], b0[t], acc[0][t]);
      acc[1][t] = wmma_bf16(a0[1], b0[t], acc[1][t]);
    }

#pragma unroll
    for (int mi = 0; mi < 2; ++mi)
      a0[mi] = load_frag(X, DM_, m0 + mi * 16, kb + 64);
#pragma unroll
    for (int t = 0; t < 4; ++t)
      b0[t] = load_frag(WT, DM_, n0 + t * 16, kb + 64);

#pragma unroll
    for (int t = 0; t < 4; ++t) {
      acc[0][t] = wmma_bf16(a1[0], b1[t], acc[0][t]);
      acc[1][t] = wmma_bf16(a1[1], b1[t], acc[1][t]);
    }
  }

  // epilogue: K-steps kb (=960) and kb+32 (=992)
#pragma unroll
  for (int mi = 0; mi < 2; ++mi)
    a1[mi] = load_frag(X, DM_, m0 + mi * 16, kb + 32);
#pragma unroll
  for (int t = 0; t < 4; ++t)
    b1[t] = load_frag(WT, DM_, n0 + t * 16, kb + 32);
#pragma unroll
  for (int t = 0; t < 4; ++t) {
    acc[0][t] = wmma_bf16(a0[0], b0[t], acc[0][t]);
    acc[1][t] = wmma_bf16(a0[1], b0[t], acc[1][t]);
  }
#pragma unroll
  for (int t = 0; t < 4; ++t) {
    acc[0][t] = wmma_bf16(a1[0], b1[t], acc[0][t]);
    acc[1][t] = wmma_bf16(a1[1], b1[t], acc[1][t]);
  }

  int lane = threadIdx.x & 31;
  int nl = lane & 15, hf = lane >> 4;
#pragma unroll
  for (int mi = 0; mi < 2; ++mi) {
#pragma unroll
    for (int t = 0; t < 4; ++t) {
      int n = n0 + t * 16 + nl;
      float bv = bias[n];
      int h = n >> 6, d = n & 63;
#pragma unroll
      for (int r = 0; r < 8; ++r) {
        int mrow = m0 + mi * 16 + r + 8 * hf;
        float v = acc[mi][t][r] + bv;
        int bb = mrow >> 11;           // / S_
        int s  = mrow & (S_ - 1);      // % S_
        if (mode == 0) {
          ((__bf16*)dst)[((size_t)(bb * H_ + h) * S_ + s) * DK_ + d] =
              (__bf16)(v * 0.125f);
        } else if (mode == 1) {
          ((__bf16*)dst)[((size_t)(bb * H_ + h) * S_ + s) * DK_ + d] = (__bf16)v;
        } else if (mode == 2) {
          ((__bf16*)dst)[((size_t)(bb * H_ + h) * DK_ + d) * S_ + s] = (__bf16)v;
        } else {
          ((float*)dst)[(size_t)mrow * DM_ + n] = v;
        }
      }
    }
  }
}

// ---------------------------------------------------------------------------
// Flash attention: one wave per (b, h, 16-query tile). Streams k in 32-wide
// steps, online softmax, causal + prefix-padding masks. V B-fragments are
// issued before the softmax block so their latency hides under exp/shuffle
// VALU work. Writes H bf16 [B*S, 1024] for the output projection.
// ---------------------------------------------------------------------------
__global__ __launch_bounds__(256, 2)
void attn_kernel(const __bf16* __restrict__ Qh,
                 const __bf16* __restrict__ Kh,
                 const __bf16* __restrict__ Vt,
                 const int* __restrict__ lens,
                 __bf16* __restrict__ Hws) {
  __shared__ alignas(16) __bf16 Plds[8 * 16 * 32];   // per-wave 16x32 P tile

  int wid = threadIdx.x >> 5;
  int gw  = blockIdx.x * 8 + wid;
  int qt  = gw & 127;          // S_/16 = 128
  int h   = (gw >> 7) & 15;
  int b   = gw >> 11;
  int q0  = qt << 4;

  int qlen = lens[b];
  int klen = lens[2 + b];

  const __bf16* Qb = Qh + (size_t)(b * H_ + h) * S_ * DK_;
  const __bf16* Kb = Kh + (size_t)(b * H_ + h) * S_ * DK_;
  const __bf16* Vb = Vt + (size_t)(b * H_ + h) * DK_ * S_;

  // Q A-fragments (16 rows x 64 d), two 32-wide K chunks, scale pre-folded.
  v16bf aq0 = load_frag(Qb, DK_, q0, 0);
  v16bf aq1 = load_frag(Qb, DK_, q0, 32);

  v8f o[4];
  float rmax[8], rsum[8];
#pragma unroll
  for (int t = 0; t < 4; ++t) o[t] = zero8();
#pragma unroll
  for (int r = 0; r < 8; ++r) { rmax[r] = -3.0e38f; rsum[r] = 0.0f; }

  int lane = threadIdx.x & 31;
  int nl = lane & 15, hf = lane >> 4;
  __bf16* pl = Plds + wid * (16 * 32);

  int klim = q0 + 16;
  if (klen < klim) klim = klen;

  for (int k0 = 0; k0 < klim; k0 += 32) {
    // scores for 16q x 32k (two 16-wide N subtiles)
    v8f s0 = zero8(), s1 = zero8();
    s0 = wmma_bf16(aq0, load_frag(Kb, DK_, k0,      0),  s0);
    s0 = wmma_bf16(aq1, load_frag(Kb, DK_, k0,      32), s0);
    s1 = wmma_bf16(aq0, load_frag(Kb, DK_, k0 + 16, 0),  s1);
    s1 = wmma_bf16(aq1, load_frag(Kb, DK_, k0 + 16, 32), s1);

    // issue V fragment loads early: independent of softmax below
    v16bf bv[4];
#pragma unroll
    for (int t = 0; t < 4; ++t) bv[t] = load_frag(Vb, S_, t * 16, k0);

    int kg0 = k0 + nl;
    int kg1 = k0 + 16 + nl;
#pragma unroll
    for (int r = 0; r < 8; ++r) {
      int qg = q0 + r + 8 * hf;
      bool qv = qg < qlen;
      float v0 = (qv && kg0 <= qg && kg0 < klen) ? s0[r] : -3.0e38f;
      float v1 = (qv && kg1 <= qg && kg1 < klen) ? s1[r] : -3.0e38f;

      float tm = fmaxf(v0, v1);
      tm = fmaxf(tm, __shfl_xor(tm, 1));
      tm = fmaxf(tm, __shfl_xor(tm, 2));
      tm = fmaxf(tm, __shfl_xor(tm, 4));
      tm = fmaxf(tm, __shfl_xor(tm, 8));

      float mnew = fmaxf(rmax[r], tm);
      float corr = (mnew > -1.0e37f) ? __expf(rmax[r] - mnew) : 1.0f;
      float p0 = (v0 > -1.0e37f) ? __expf(v0 - mnew) : 0.0f;
      float p1 = (v1 > -1.0e37f) ? __expf(v1 - mnew) : 0.0f;
      rmax[r] = mnew;

      float ts = p0 + p1;
      ts += __shfl_xor(ts, 1);
      ts += __shfl_xor(ts, 2);
      ts += __shfl_xor(ts, 4);
      ts += __shfl_xor(ts, 8);
      rsum[r] = rsum[r] * corr + ts;

      o[0][r] *= corr; o[1][r] *= corr; o[2][r] *= corr; o[3][r] *= corr;

      // C-layout -> LDS (row-major 16x32) for A-fragment reload
      pl[(r + 8 * hf) * 32 + nl]      = (__bf16)p0;
      pl[(r + 8 * hf) * 32 + 16 + nl] = (__bf16)p1;
    }

    // O += P(16x32) x V(32k x 64d); same-wave DS ops are in-order, no barrier.
    v16bf ap = load_frag(pl, 32, 0, 0);
#pragma unroll
    for (int t = 0; t < 4; ++t) o[t] = wmma_bf16(ap, bv[t], o[t]);
  }

  // normalize; fully masked rows (rsum==0) -> 0 (nan_to_num semantics)
#pragma unroll
  for (int t = 0; t < 4; ++t) {
#pragma unroll
    for (int r = 0; r < 8; ++r) {
      int qg = q0 + r + 8 * hf;
      float den = rsum[r];
      float val = (den > 0.0f) ? (o[t][r] / den) : 0.0f;
      Hws[((size_t)(b * S_ + qg)) * DM_ + h * DK_ + t * 16 + nl] = (__bf16)val;
    }
  }
}

// ---------------------------------------------------------------------------
// Host-side orchestration
// ---------------------------------------------------------------------------
extern "C" void kernel_launch(void* const* d_in, const int* in_sizes, int n_in,
                              void* d_out, int out_size, void* d_ws,
                              size_t ws_size, hipStream_t stream) {
  (void)in_sizes; (void)n_in; (void)out_size; (void)ws_size;

  const float* in_Q  = (const float*)d_in[0];
  const float* in_K  = (const float*)d_in[1];
  const float* in_V  = (const float*)d_in[2];
  const int*   qmask = (const int*)d_in[3];
  const int*   kmask = (const int*)d_in[4];
  const float* W_Q = (const float*)d_in[5];  const float* b_Q = (const float*)d_in[6];
  const float* W_K = (const float*)d_in[7];  const float* b_K = (const float*)d_in[8];
  const float* W_V = (const float*)d_in[9];  const float* b_V = (const float*)d_in[10];
  const float* W_O = (const float*)d_in[11]; const float* b_O = (const float*)d_in[12];

  // workspace partition (256B aligned)
  char* w = (char*)d_ws;
  auto carve = [&](size_t bytes) -> char* {
    char* p = w;
    w += (bytes + 255) & ~(size_t)255;
    return p;
  };
  const size_t actBytes  = (size_t)MROWS_ * DM_ * sizeof(__bf16);       // 8 MB
  const size_t headBytes = (size_t)B_ * H_ * S_ * DK_ * sizeof(__bf16); // 8 MB
  const size_t wtBytes   = (size_t)DM_ * DM_ * sizeof(__bf16);          // 2 MB

  __bf16* XQ  = (__bf16*)carve(actBytes);
  __bf16* XK  = (__bf16*)carve(actBytes);
  __bf16* XV  = (__bf16*)carve(actBytes);
  __bf16* WTQ = (__bf16*)carve(wtBytes);
  __bf16* WTK = (__bf16*)carve(wtBytes);
  __bf16* WTV = (__bf16*)carve(wtBytes);
  __bf16* WTO = (__bf16*)carve(wtBytes);
  __bf16* Qh  = (__bf16*)carve(headBytes);
  __bf16* Kh  = (__bf16*)carve(headBytes);
  __bf16* Vt  = (__bf16*)carve(headBytes);
  __bf16* Hws = (__bf16*)carve(actBytes);
  int*    lens = (int*)carve(4 * sizeof(int));

  // 1) pack activations to bf16
  const int n4 = (MROWS_ * DM_) / 4;  // 1M vec4 elems
  dim3 cvtGrid((n4 + 255) / 256), cvtBlk(256);
  cvt_f32_bf16<<<cvtGrid, cvtBlk, 0, stream>>>(in_Q, XQ, n4);
  cvt_f32_bf16<<<cvtGrid, cvtBlk, 0, stream>>>(in_K, XK, n4);
  cvt_f32_bf16<<<cvtGrid, cvtBlk, 0, stream>>>(in_V, XV, n4);

  // 2) transpose weights to N x K bf16
  dim3 trGrid(32, 32), trBlk(32, 8);
  transpose_w_bf16<<<trGrid, trBlk, 0, stream>>>(W_Q, WTQ);
  transpose_w_bf16<<<trGrid, trBlk, 0, stream>>>(W_K, WTK);
  transpose_w_bf16<<<trGrid, trBlk, 0, stream>>>(W_V, WTV);
  transpose_w_bf16<<<trGrid, trBlk, 0, stream>>>(W_O, WTO);

  // 3) padding lengths
  compute_lens<<<dim3(B_, 2), 256, 0, stream>>>(qmask, kmask, lens);

  // 4) Q/K/V projections (WMMA), 32x64 per wave, pipelined
  dim3 gGrid(MROWS_ / 256, DM_ / 64), gBlk(256);
  gemm_bf16<<<gGrid, gBlk, 0, stream>>>(XQ, WTQ, b_Q, Qh, 0);
  gemm_bf16<<<gGrid, gBlk, 0, stream>>>(XK, WTK, b_K, Kh, 1);
  gemm_bf16<<<gGrid, gBlk, 0, stream>>>(XV, WTV, b_V, Vt, 2);

  // 5) flash attention: 4096 waves = 512 blocks x 8 waves
  attn_kernel<<<dim3((B_ * H_ * (S_ / 16)) / 8), 256, 0, stream>>>(
      Qh, Kh, Vt, lens, Hws);

  // 6) output projection -> fp32 d_out
  gemm_bf16<<<gGrid, gBlk, 0, stream>>>(Hws, WTO, b_O, d_out, 3);
}
